// Attention_10496900072088
// MI455X (gfx1250) — matrix operand order repaired and checked by reading, not compiled, not run on hardware
//
#include <hip/hip_runtime.h>
#include <hip/hip_bf16.h>
#include <hip/hip_fp16.h>

#define DIM       768
#define HEADS     12
#define HEAD_DIM  64
#define BATCH     4
#define SEQ       2048
#define ROWS      (BATCH * SEQ)   // 8192

typedef _Float16 half8 __attribute__((ext_vector_type(8)));
typedef _Float16 v16h  __attribute__((ext_vector_type(16)));
typedef float    v8f   __attribute__((ext_vector_type(8)));
typedef int      v4i   __attribute__((ext_vector_type(4)));

// ---------------------------------------------------------------------------
// CDNA5 async global->LDS copy (ASYNCcnt) with compile-safe fallback.
// Builtin signature (from hipcc diagnostic): (v4i AS1*, v4i AS3*, Ii, Ii).
// ---------------------------------------------------------------------------
#if defined(__has_builtin)
#  if __has_builtin(__builtin_amdgcn_global_load_async_to_lds_b128) && \
      __has_builtin(__builtin_amdgcn_s_wait_asynccnt)
#    define USE_ASYNC_LDS 1
#  endif
#endif
#ifndef USE_ASYNC_LDS
#  define USE_ASYNC_LDS 0
#endif

#if USE_ASYNC_LDS
#  define WAIT_ASYNC(n) __builtin_amdgcn_s_wait_asynccnt(n)
#else
#  define WAIT_ASYNC(n) ((void)0)
#endif

// copy 16 bytes (8 halfs) from global to LDS
__device__ inline void async_cp16(const _Float16* src, _Float16* dst_lds) {
#if USE_ASYNC_LDS
  __builtin_amdgcn_global_load_async_to_lds_b128(
      (__attribute__((address_space(1))) v4i*)(uintptr_t)src,
      (__attribute__((address_space(3))) v4i*)(uint32_t)(uintptr_t)dst_lds,
      /*offset=*/0, /*cpol=*/0);
#else
  *(half8*)dst_lds = *(const half8*)src;
#endif
}

// ---------------------------------------------------------------------------
// WMMA fragment helpers (wave32, v_wmma_f32_16x16x32_f16)
//
// A (16x32 f16, MxK): lane L holds row M=L%16; elements 0..7 -> K = 8*(L/16)+e,
//                     elements 8..15 -> K = 16 + 8*(L/16) + (e-8).
// B (32x16 f16, KxN): lane L holds column N=L%16; same K order as A.
//                     => load contiguous from B^T (N-major) memory.
// C/D (16x16 f32):    lane L, reg r -> M = r + 8*(L/16), N = L%16.
// ---------------------------------------------------------------------------

__device__ inline v16h frag_a_load(const _Float16* tile, int lda, int lane) {
  const int m  = lane & 15;
  const int kb = (lane >> 4) << 3;              // 0 or 8
  const _Float16* rp = tile + m * lda + kb;
  half8 lo = *(const half8*)(rp);               // K = kb .. kb+7
  half8 hi = *(const half8*)(rp + 16);          // K = 16+kb .. 16+kb+7
  v16h r;
#pragma unroll
  for (int i = 0; i < 8; ++i) { r[i] = lo[i]; r[i + 8] = hi[i]; }
  return r;
}

__device__ inline v16h frag_b_load(const _Float16* tileT, int ldb, int lane) {
  const int n  = lane & 15;
  const int kb = (lane >> 4) << 3;              // 0 or 8
  const _Float16* rp = tileT + n * ldb + kb;
  half8 lo = *(const half8*)(rp);
  half8 hi = *(const half8*)(rp + 16);
  v16h r;
#pragma unroll
  for (int i = 0; i < 8; ++i) { r[i] = lo[i]; r[i + 8] = hi[i]; }
  return r;
}

__device__ inline v8f wmma16(v16h a, v16h b, v8f c) {
  return __builtin_amdgcn_wmma_f32_16x16x32_f16(
      /*neg_a=*/false, a, /*neg_b=*/false, b,
      /*c_mod=*/(short)0, c, /*reuse_a=*/false, /*reuse_b=*/false);
}

// ---------------------------------------------------------------------------
// Precision conversion / weight transposition
// ---------------------------------------------------------------------------

__global__ void k_f32_to_f16(const float* __restrict__ in,
                             _Float16* __restrict__ out, int n) {
  int i = blockIdx.x * blockDim.x + threadIdx.x;
  if (i < n) out[i] = (_Float16)in[i];
}

__global__ void k_transpose_f32_to_f16(const float* __restrict__ in,
                                       _Float16* __restrict__ out,
                                       int R, int C) {
  int i = blockIdx.x * blockDim.x + threadIdx.x;
  if (i >= R * C) return;
  int r = i / C, c = i - r * C;
  out[(size_t)c * R + r] = (_Float16)in[i];    // out = in^T, [C x R]
}

// ---------------------------------------------------------------------------
// QKV GEMM: [8192x768] x [768x2304] + bias. All 8 waves of a block share the
// same 64-column B tile -> stage B through LDS with double-buffered async
// copies. Each wave computes a 32(M)x64(N) tile: 8 WMMAs per K=32 step.
// Epilogue scatters into q:[B,H,N,D], k:[B,H,N,D], vT:[B,H,D,N] (f16).
// Grid is sized exactly; no early returns (barriers inside the loop).
// ---------------------------------------------------------------------------

__global__ __launch_bounds__(256) void k_qkv_gemm(
    const _Float16* __restrict__ xh, const _Float16* __restrict__ wT,
    const float* __restrict__ bias,
    _Float16* __restrict__ qb, _Float16* __restrict__ kb,
    _Float16* __restrict__ vT) {
  __shared__ _Float16 Bs[2][64 * 32];          // 2 x 4 KB
  const int tid  = threadIdx.x;
  const int lane = tid & 31;
  const int wv   = tid >> 5;
  const int wid  = blockIdx.x * 8 + wv;
  const int MT   = ROWS / 32;                  // 256 (8 | 256 -> block shares nt)
  const int mt   = wid % MT, nt = wid / MT;
  const int m0   = mt * 32,  n0 = nt * 64;
  const int NK   = DIM / 32;                   // 24

  // staging assignment: thread -> (B row, 8-half column chunk)
  const int srow = tid >> 2;                   // 0..63
  const int scol = (tid & 3) << 3;             // 0,8,16,24
  const _Float16* wrow = wT + (size_t)(n0 + srow) * DIM + scol;
  _Float16* bd0 = &Bs[0][srow * 32 + scol];
  _Float16* bd1 = &Bs[1][srow * 32 + scol];

  async_cp16(wrow, bd0);                       // k0 = 0 into buffer 0

  v8f acc[2][4] = {};
  const _Float16* A = xh + (size_t)m0 * DIM;
  for (int i = 0; i < NK; ++i) {
    const int cur = i & 1;
    if (i + 1 < NK) {
      async_cp16(wrow + (i + 1) * 32, cur ? bd0 : bd1);
      WAIT_ASYNC(1);                           // previous stage has landed
    } else {
      WAIT_ASYNC(0);
    }
    __syncthreads();                           // publish staged B tile
    const int k0 = i * 32;
    v16h a0 = frag_a_load(A + k0,            DIM, lane);
    v16h a1 = frag_a_load(A + 16 * DIM + k0, DIM, lane);
#pragma unroll
    for (int j = 0; j < 4; ++j) {
      v16h b = frag_b_load(&Bs[cur][j * 16 * 32], 32, lane);
      acc[0][j] = wmma16(a0, b, acc[0][j]);
      acc[1][j] = wmma16(a1, b, acc[1][j]);
    }
    __syncthreads();                           // done reading before overwrite
  }

  const int hi = lane >> 4, nlo = lane & 15;
#pragma unroll
  for (int j = 0; j < 4; ++j) {
    const int c    = n0 + j * 16 + nlo;
    const int part = c / DIM;                  // 0=q 1=k 2=v
    const int rem  = c - part * DIM;
    const int h    = rem / HEAD_DIM;
    const int d    = rem - h * HEAD_DIM;
    const float bv = bias[c];
#pragma unroll
    for (int i = 0; i < 2; ++i) {
#pragma unroll
      for (int r = 0; r < 8; ++r) {
        const int m    = m0 + i * 16 + r + 8 * hi;
        const int bidx = m / SEQ;
        const int n    = m - bidx * SEQ;
        const int bh   = bidx * HEADS + h;
        const _Float16 hv = (_Float16)(acc[i][j][r] + bv);
        if (part == 0)      qb[((size_t)bh * SEQ + n) * HEAD_DIM + d] = hv;
        else if (part == 1) kb[((size_t)bh * SEQ + n) * HEAD_DIM + d] = hv;
        else                vT[((size_t)bh * HEAD_DIM + d) * SEQ + n] = hv;
      }
    }
  }
}

// ---------------------------------------------------------------------------
// Flash attention: 8 waves per block own 8 consecutive 16-row q tiles of the
// SAME (b,h), so the K/V stream is block-shared: stage 64-key K and V^T
// chunks (8 KB each) through LDS with double-buffered async copies.
// Per 64-key chunk: 8 QK WMMAs + 8 PV WMMAs; online softmax amortized.
// P tile round-trips through per-wave LDS (C-frag -> A-frag layout).
// ---------------------------------------------------------------------------

__global__ __launch_bounds__(256) void k_attention(
    const _Float16* __restrict__ qb, const _Float16* __restrict__ kb,
    const _Float16* __restrict__ vT, _Float16* __restrict__ ctx) {
  __shared__ _Float16 Ks[2][64 * 64];          // 2 x 8 KB, rows = key, cols = d
  __shared__ _Float16 Vs[2][64 * 64];          // 2 x 8 KB, rows = d, cols = key
  __shared__ _Float16 Pb[8][16 * 64];          // 2 KB per wave
  const int tid  = threadIdx.x;
  const int lane = tid & 31;
  const int w    = tid >> 5;
  const int bh   = blockIdx.x / 16;            // 16 blocks per (b,h)
  const int q0   = ((blockIdx.x % 16) * 8 + w) * 16;
  const float scale = 0.125f;                  // 1/sqrt(64)

  // staging assignment: thread -> (row 0..63, 16-half column chunk)
  const int srow = tid >> 2;
  const int scol = (tid & 3) << 4;             // 0,16,32,48
  const _Float16* kbase = kb + (size_t)bh * SEQ * HEAD_DIM;
  const _Float16* vbase = vT + (size_t)bh * HEAD_DIM * SEQ;

  const _Float16* qtile = qb + ((size_t)bh * SEQ + q0) * HEAD_DIM;
  const v16h qa0 = frag_a_load(qtile,      HEAD_DIM, lane);   // d 0..31
  const v16h qa1 = frag_a_load(qtile + 32, HEAD_DIM, lane);   // d 32..63

  v8f o[4] = {};
  float mrow[8], lrow[8];
#pragma unroll
  for (int r = 0; r < 8; ++r) { mrow[r] = -3.0e38f; lrow[r] = 0.f; }

  _Float16* P = Pb[w];
  const int hi = lane >> 4, nlo = lane & 15;
  const int NKT = SEQ / 64;                    // 32 chunks

  // stage chunk 0 into buffer 0 (4 async instructions per wave)
  {
    const _Float16* ks = kbase + (size_t)srow * HEAD_DIM + scol;
    const _Float16* vs = vbase + (size_t)srow * SEQ + scol;
    async_cp16(ks,     &Ks[0][srow * 64 + scol]);
    async_cp16(ks + 8, &Ks[0][srow * 64 + scol + 8]);
    async_cp16(vs,     &Vs[0][srow * 64 + scol]);
    async_cp16(vs + 8, &Vs[0][srow * 64 + scol + 8]);
  }

  for (int it = 0; it < NKT; ++it) {
    const int cur = it & 1;
    if (it + 1 < NKT) {
      const int ktn = (it + 1) * 64;
      const _Float16* ks = kbase + (size_t)(ktn + srow) * HEAD_DIM + scol;
      const _Float16* vs = vbase + (size_t)srow * SEQ + ktn + scol;
      _Float16* kd = &Ks[cur ^ 1][srow * 64 + scol];
      _Float16* vd = &Vs[cur ^ 1][srow * 64 + scol];
      async_cp16(ks,     kd);
      async_cp16(ks + 8, kd + 8);
      async_cp16(vs,     vd);
      async_cp16(vs + 8, vd + 8);
      WAIT_ASYNC(4);                           // previous chunk has landed
    } else {
      WAIT_ASYNC(0);
    }
    __syncthreads();                           // publish staged K/V chunk

    // S = Q K^T for 64 keys: four 16x16 score tiles, 2 WMMAs each (D=64)
    v8f s[4];
#pragma unroll
    for (int i = 0; i < 4; ++i) {
      v8f z = {};
      v16h b0 = frag_b_load(&Ks[cur][(i * 16) * 64],      64, lane);
      v16h b1 = frag_b_load(&Ks[cur][(i * 16) * 64 + 32], 64, lane);
      z = wmma16(qa0, b0, z);
      z = wmma16(qa1, b1, z);
      s[i] = z;
    }
    // online softmax over the 64 new keys (each row lives across 16 lanes)
#pragma unroll
    for (int r = 0; r < 8; ++r) {
      float a0 = s[0][r] * scale;
      float a1 = s[1][r] * scale;
      float a2 = s[2][r] * scale;
      float a3 = s[3][r] * scale;
      float loc = fmaxf(fmaxf(a0, a1), fmaxf(a2, a3));
#pragma unroll
      for (int mask = 1; mask < 16; mask <<= 1)
        loc = fmaxf(loc, __shfl_xor(loc, mask, 32));
      const float mnew  = fmaxf(mrow[r], loc);
      const float alpha = __expf(mrow[r] - mnew);
      const float p0 = __expf(a0 - mnew);
      const float p1 = __expf(a1 - mnew);
      const float p2 = __expf(a2 - mnew);
      const float p3 = __expf(a3 - mnew);
      float ps = (p0 + p1) + (p2 + p3);
#pragma unroll
      for (int mask = 1; mask < 16; mask <<= 1)
        ps += __shfl_xor(ps, mask, 32);
      lrow[r] = lrow[r] * alpha + ps;
      mrow[r] = mnew;
#pragma unroll
      for (int j = 0; j < 4; ++j) o[j][r] = o[j][r] * alpha;
      const int m = r + 8 * hi;
      P[m * 64 + nlo]      = (_Float16)p0;
      P[m * 64 + 16 + nlo] = (_Float16)p1;
      P[m * 64 + 32 + nlo] = (_Float16)p2;
      P[m * 64 + 48 + nlo] = (_Float16)p3;
    }
    // O += P[16x64] @ V[64x64]  (V^T rows are d, contiguous in k)
    v16h pa0 = frag_a_load(P,      64, lane);  // k 0..31
    v16h pa1 = frag_a_load(P + 32, 64, lane);  // k 32..63
#pragma unroll
    for (int j = 0; j < 4; ++j) {
      const _Float16* vrows = &Vs[cur][(j * 16) * 64];
      v16h b0 = frag_b_load(vrows,      64, lane);
      v16h b1 = frag_b_load(vrows + 32, 64, lane);
      o[j] = wmma16(pa0, b0, o[j]);
      o[j] = wmma16(pa1, b1, o[j]);
    }
    __syncthreads();                           // done reading before overwrite
  }

  // normalize and write ctx [B, N, 768] f16
  const int bidx = bh / HEADS;
  const int h    = bh - bidx * HEADS;
#pragma unroll
  for (int j = 0; j < 4; ++j) {
#pragma unroll
    for (int r = 0; r < 8; ++r) {
      const int m = q0 + r + 8 * hi;
      const float val = o[j][r] / lrow[r];
      ctx[((size_t)bidx * SEQ + m) * DIM + h * HEAD_DIM + j * 16 + nlo] =
          (_Float16)val;
    }
  }
}

// ---------------------------------------------------------------------------
// Output projection: [8192x768] x [768x768] + bias -> f32 d_out.
// Same LDS-staged structure as the QKV GEMM.
// ---------------------------------------------------------------------------

__global__ __launch_bounds__(256) void k_proj_gemm(
    const _Float16* __restrict__ ctx, const _Float16* __restrict__ wT,
    const float* __restrict__ bias, float* __restrict__ out) {
  __shared__ _Float16 Bs[2][64 * 32];
  const int tid  = threadIdx.x;
  const int lane = tid & 31;
  const int wv   = tid >> 5;
  const int wid  = blockIdx.x * 8 + wv;
  const int MT   = ROWS / 32;                  // 256
  const int mt   = wid % MT, nt = wid / MT;
  const int m0   = mt * 32,  n0 = nt * 64;
  const int NK   = DIM / 32;                   // 24

  const int srow = tid >> 2;
  const int scol = (tid & 3) << 3;
  const _Float16* wrow = wT + (size_t)(n0 + srow) * DIM + scol;
  _Float16* bd0 = &Bs[0][srow * 32 + scol];
  _Float16* bd1 = &Bs[1][srow * 32 + scol];

  async_cp16(wrow, bd0);

  v8f acc[2][4] = {};
  const _Float16* A = ctx + (size_t)m0 * DIM;
  for (int i = 0; i < NK; ++i) {
    const int cur = i & 1;
    if (i + 1 < NK) {
      async_cp16(wrow + (i + 1) * 32, cur ? bd0 : bd1);
      WAIT_ASYNC(1);
    } else {
      WAIT_ASYNC(0);
    }
    __syncthreads();
    const int k0 = i * 32;
    v16h a0 = frag_a_load(A + k0,            DIM, lane);
    v16h a1 = frag_a_load(A + 16 * DIM + k0, DIM, lane);
#pragma unroll
    for (int j = 0; j < 4; ++j) {
      v16h b = frag_b_load(&Bs[cur][j * 16 * 32], 32, lane);
      acc[0][j] = wmma16(a0, b, acc[0][j]);
      acc[1][j] = wmma16(a1, b, acc[1][j]);
    }
    __syncthreads();
  }

  const int hi = lane >> 4, nlo = lane & 15;
#pragma unroll
  for (int j = 0; j < 4; ++j) {
    const int c = n0 + j * 16 + nlo;
    const float bv = bias[c];
#pragma unroll
    for (int i = 0; i < 2; ++i) {
#pragma unroll
      for (int r = 0; r < 8; ++r) {
        const int m = m0 + i * 16 + r + 8 * hi;
        out[(size_t)m * DIM + c] = acc[i][j][r] + bv;
      }
    }
  }
}

// ---------------------------------------------------------------------------
// Host launcher
// ---------------------------------------------------------------------------

extern "C" void kernel_launch(void* const* d_in, const int* in_sizes, int n_in,
                              void* d_out, int out_size, void* d_ws,
                              size_t ws_size, hipStream_t stream) {
  (void)in_sizes; (void)n_in; (void)out_size; (void)ws_size;
  const float* x      = (const float*)d_in[0];
  const float* w_qkv  = (const float*)d_in[1];
  const float* b_qkv  = (const float*)d_in[2];
  const float* w_proj = (const float*)d_in[3];
  const float* b_proj = (const float*)d_in[4];
  float* out = (float*)d_out;

  char* p = (char*)d_ws;
  auto carve = [&](size_t bytes) -> void* {
    void* q = (void*)p;
    p += (bytes + 255) & ~(size_t)255;
    return q;
  };
  _Float16* xh     = (_Float16*)carve((size_t)ROWS * DIM * 2);
  _Float16* wqkvT  = (_Float16*)carve((size_t)3 * DIM * DIM * 2);
  _Float16* wprojT = (_Float16*)carve((size_t)DIM * DIM * 2);
  _Float16* qbuf   = (_Float16*)carve((size_t)ROWS * DIM * 2);
  _Float16* kbuf   = (_Float16*)carve((size_t)ROWS * DIM * 2);
  _Float16* vTbuf  = (_Float16*)carve((size_t)ROWS * DIM * 2);
  _Float16* ctx    = (_Float16*)carve((size_t)ROWS * DIM * 2);

  int n = ROWS * DIM;
  k_f32_to_f16<<<(n + 255) / 256, 256, 0, stream>>>(x, xh, n);
  n = DIM * 3 * DIM;
  k_transpose_f32_to_f16<<<(n + 255) / 256, 256, 0, stream>>>(w_qkv, wqkvT,
                                                              DIM, 3 * DIM);
  n = DIM * DIM;
  k_transpose_f32_to_f16<<<(n + 255) / 256, 256, 0, stream>>>(w_proj, wprojT,
                                                              DIM, DIM);

  // QKV: 256 x 36 wave-tiles, 8 waves/block (exact grid, no early returns)
  k_qkv_gemm<<<(256 * 36) / 8, 256, 0, stream>>>(xh, wqkvT, b_qkv,
                                                 qbuf, kbuf, vTbuf);
  // Attention: 48 bh * 16 blocks each
  k_attention<<<BATCH * HEADS * 16, 256, 0, stream>>>(qbuf, kbuf, vTbuf, ctx);
  // Projection: 256 x 12 wave-tiles
  k_proj_gemm<<<(256 * 12) / 8, 256, 0, stream>>>(ctx, wprojT, b_proj, out);
}
